// CasualAttention_47167330845112
// MI455X (gfx1250) — compile-verified, matrix operand order checked by reading, not run on hardware
//
#include <hip/hip_runtime.h>
#include <hip/hip_bf16.h>

// ---------------------------------------------------------------------------
// Causal attention block for MI455X (gfx1250, wave32, WMMA bf16->f32).
// B=4, L=2048, C=1024, H=16, D=64.
// Pipeline: cvt -> qkv GEMM (WMMA) -> flash attention (WMMA) -> proj GEMM.
// ---------------------------------------------------------------------------

#define BB 4
#define LL 2048
#define CCH 1024
#define HH 16
#define DD 64

typedef __bf16 bh16;
typedef __attribute__((ext_vector_type(16))) __bf16 v16bf;
typedef __attribute__((ext_vector_type(8)))  float  v8f;
typedef __attribute__((ext_vector_type(4)))  unsigned int u32x4;

union Frag32B { v16bf v; u32x4 c[2]; };

// D = A(16x32)*B(32x16) + C, f32 accumulate.  RA = reuse-A operand hint
// (legal when previous instruction is the identical WMMA with same A).
#define WMMA(A_, B_, C_, RA_) \
  __builtin_amdgcn_wmma_f32_16x16x32_bf16(false, (A_), false, (B_), (short)0, (C_), (RA_), false)

__device__ __forceinline__ bh16 f2bf(float f) {
  unsigned u = __builtin_bit_cast(unsigned, f);
  u = (u + 0x7FFFu + ((u >> 16) & 1u)) >> 16;
  unsigned short s = (unsigned short)u;
  return __builtin_bit_cast(bh16, s);
}

__device__ __forceinline__ v8f vzero8() {
  v8f z;
#pragma unroll
  for (int i = 0; i < 8; ++i) z[i] = 0.0f;
  return z;
}

// Load a 16x32 bf16 A-fragment (or B-fragment from a transposed matrix) from a
// row-major source.  Per ISA 16-bit A layout: lane = {m = lane&15, khalf =
// lane>>4}; VGPRs 0..3 hold K = khalf*8 + 0..7, VGPRs 4..7 hold K = 16 +
// khalf*8 + 0..7 -> two contiguous 16-byte chunks per lane.
__device__ __forceinline__ v16bf load_frag(const bh16* __restrict__ base, int ld) {
  const int lane = threadIdx.x & 31;
  const int m  = lane & 15;
  const int kh = lane >> 4;
  const bh16* p = base + (size_t)m * ld + kh * 8;
  Frag32B f;
  f.c[0] = *(const u32x4*)(p);
  f.c[1] = *(const u32x4*)(p + 16);
  return f.v;
}

// ---------------------------------------------------------------------------
// Conversion kernels
// ---------------------------------------------------------------------------
__global__ void k_cvt(const float* __restrict__ in, bh16* __restrict__ out, int n) {
  int i = blockIdx.x * blockDim.x + threadIdx.x;
  int stride = gridDim.x * blockDim.x;
  for (; i < n; i += stride) out[i] = f2bf(in[i]);
}

// out[n*K + k] = in[k*N + n]   (store W transposed so B-fragments are contiguous)
__global__ void k_cvt_t(const float* __restrict__ in, bh16* __restrict__ out,
                        int K, int N) {
  int i = blockIdx.x * blockDim.x + threadIdx.x;
  int stride = gridDim.x * blockDim.x;
  int total = K * N;
  for (; i < total; i += stride) {
    int n = i / K;
    int k = i - n * K;
    out[i] = f2bf(in[(size_t)k * N + n]);
  }
}

// ---------------------------------------------------------------------------
// QKV GEMM: qkv = xb[8192,1024] @ W_attn[1024,3072]  (W stored transposed).
// One wave owns a 16x64 output tile; K-loop is register double-buffered so
// global_load_b128 of step k+1 overlaps the 4 WMMAs of step k.
// Q,K scattered to [B,H,L,D] bf16; V scattered transposed to [B,H,D,L] bf16.
// ---------------------------------------------------------------------------
__global__ __launch_bounds__(256) void k_qkv_gemm(
    const bh16* __restrict__ xb, const bh16* __restrict__ wT,
    bh16* __restrict__ Qb, bh16* __restrict__ Kb, bh16* __restrict__ Vt) {
  const int wave = threadIdx.x >> 5;
  const int lane = threadIdx.x & 31;
  const int mrow0 = (blockIdx.y * 8 + wave) * 16;
  const int ntb = blockIdx.x * 64;

  const bh16* ap = xb + (size_t)mrow0 * CCH;
  const bh16* bp = wT + (size_t)ntb * CCH;

  v8f acc[4];
#pragma unroll
  for (int t = 0; t < 4; ++t) acc[t] = vzero8();

  // prologue: fragments for kk = 0
  v16bf a  = load_frag(ap, CCH);
  v16bf b0 = load_frag(bp + 0 * 16 * CCH, CCH);
  v16bf b1 = load_frag(bp + 1 * 16 * CCH, CCH);
  v16bf b2 = load_frag(bp + 2 * 16 * CCH, CCH);
  v16bf b3 = load_frag(bp + 3 * 16 * CCH, CCH);

  for (int kk = 32; kk < CCH; kk += 32) {
    // prefetch next K-step while the WMMAs below consume the current one
    v16bf an  = load_frag(ap + kk, CCH);
    v16bf bn0 = load_frag(bp + 0 * 16 * CCH + kk, CCH);
    v16bf bn1 = load_frag(bp + 1 * 16 * CCH + kk, CCH);
    v16bf bn2 = load_frag(bp + 2 * 16 * CCH + kk, CCH);
    v16bf bn3 = load_frag(bp + 3 * 16 * CCH + kk, CCH);

    acc[0] = WMMA(a, b0, acc[0], false);
    acc[1] = WMMA(a, b1, acc[1], true);   // same A as previous WMMA
    acc[2] = WMMA(a, b2, acc[2], true);
    acc[3] = WMMA(a, b3, acc[3], true);

    a = an; b0 = bn0; b1 = bn1; b2 = bn2; b3 = bn3;
  }
  // epilogue
  acc[0] = WMMA(a, b0, acc[0], false);
  acc[1] = WMMA(a, b1, acc[1], true);
  acc[2] = WMMA(a, b2, acc[2], true);
  acc[3] = WMMA(a, b3, acc[3], true);

  // C/D layout: lanes 0-15 -> N=lane, M=v;  lanes 16-31 -> N=lane-16, M=8+v.
  const int mo = (lane >> 4) * 8;
  const int nl = lane & 15;
#pragma unroll
  for (int t = 0; t < 4; ++t) {
    int ng = ntb + t * 16 + nl;        // column in [0,3072)
    int sec = ng >> 10;                // 0=Q 1=K 2=V
    int cc = ng & (CCH - 1);
    int h = cc >> 6, d = cc & (DD - 1);
#pragma unroll
    for (int v = 0; v < 8; ++v) {
      int row = mrow0 + mo + v;        // in [0, B*L)
      int b = row >> 11, l = row & (LL - 1);
      bh16 val = f2bf(acc[t][v]);
      size_t hb = (size_t)(b * HH + h);
      if (sec == 0)      Qb[(hb * LL + l) * DD + d] = val;
      else if (sec == 1) Kb[(hb * LL + l) * DD + d] = val;
      else               Vt[(hb * DD + d) * LL + l] = val;
    }
  }
}

// ---------------------------------------------------------------------------
// Flash-style causal attention.  One wave per 16-row Q tile; key blocks of 32.
// V fragments + next K fragments are issued before the softmax VALU so their
// latency hides under the exp/shuffle work.
// ---------------------------------------------------------------------------
__global__ __launch_bounds__(128) void k_attn(
    const bh16* __restrict__ Qb, const bh16* __restrict__ Kb,
    const bh16* __restrict__ Vt, bh16* __restrict__ yb) {
  __shared__ bh16 plds[4 * 16 * 32];   // per-wave 16x32 P tile (C->A transpose)

  const int wave = threadIdx.x >> 5;
  const int lane = threadIdx.x & 31;
  const int bh = blockIdx.y;
  const int b = bh >> 4, h = bh & (HH - 1);
  const int qbase = (blockIdx.x * 4 + wave) * 16;

  const bh16* Qh = Qb + (size_t)bh * LL * DD;
  const bh16* Kh = Kb + (size_t)bh * LL * DD;
  const bh16* Vh = Vt + (size_t)bh * DD * LL;
  bh16* ybp = yb + (size_t)b * LL * CCH;

  // Q fragments for d=0..31 and d=32..63 (reused across all key blocks).
  v16bf qa0 = load_frag(Qh + (size_t)qbase * DD, DD);
  v16bf qa1 = load_frag(Qh + (size_t)qbase * DD + 32, DD);

  v8f o[4];
#pragma unroll
  for (int t = 0; t < 4; ++t) o[t] = vzero8();
  float mrow[8], lrow[8];
#pragma unroll
  for (int v = 0; v < 8; ++v) { mrow[v] = -__builtin_inff(); lrow[v] = 0.0f; }

  const float scale = 0.03125f;  // 1/sqrt(C) = 1/32 (reference scales by C!)
  const int mo = (lane >> 4) * 8;
  const int nl = lane & 15;
  bh16* pl = &plds[wave * 512];

  // prologue: K fragments for kb = 0  (jt = key subtile, s = d-half)
  v16bf k00 = load_frag(Kh + 0, DD);
  v16bf k01 = load_frag(Kh + 32, DD);
  v16bf k10 = load_frag(Kh + (size_t)16 * DD, DD);
  v16bf k11 = load_frag(Kh + (size_t)16 * DD + 32, DD);

  for (int kb = 0; kb <= qbase; kb += 32) {
    // ---- S = Q @ K^T for 32 keys: two 16-col C fragments, 4 WMMAs ----
    v8f s0 = vzero8(), s1 = vzero8();
    s0 = WMMA(qa0, k00, s0, false);
    s0 = WMMA(qa1, k01, s0, false);
    s1 = WMMA(qa0, k10, s1, false);
    s1 = WMMA(qa1, k11, s1, false);

    // ---- issue V loads (used after softmax) and next-K prefetch now ----
    v16bf vb0 = load_frag(Vh + (size_t)(0 * 16) * LL + kb, LL);
    v16bf vb1 = load_frag(Vh + (size_t)(1 * 16) * LL + kb, LL);
    v16bf vb2 = load_frag(Vh + (size_t)(2 * 16) * LL + kb, LL);
    v16bf vb3 = load_frag(Vh + (size_t)(3 * 16) * LL + kb, LL);
    if (kb + 32 <= qbase) {            // wave-uniform branch
      const bh16* kp = Kh + (size_t)(kb + 32) * DD;
      k00 = load_frag(kp, DD);
      k01 = load_frag(kp + 32, DD);
      k10 = load_frag(kp + (size_t)16 * DD, DD);
      k11 = load_frag(kp + (size_t)16 * DD + 32, DD);
    }

    // ---- causal mask + online softmax (row = mo+v, col = nl per C layout) --
    float p0[8], p1[8];
#pragma unroll
    for (int v = 0; v < 8; ++v) {
      int qrow = qbase + mo + v;
      float x0 = s0[v] * scale;
      float x1 = s1[v] * scale;
      if (kb + nl > qrow)      x0 = -__builtin_inff();
      if (kb + 16 + nl > qrow) x1 = -__builtin_inff();
      float rm = fmaxf(x0, x1);
#pragma unroll
      for (int sh = 1; sh <= 8; sh <<= 1) rm = fmaxf(rm, __shfl_xor(rm, sh));
      float mnew = fmaxf(mrow[v], rm);
      float corr = __expf(mrow[v] - mnew);
      float e0 = __expf(x0 - mnew);
      float e1 = __expf(x1 - mnew);
      float rs = e0 + e1;
#pragma unroll
      for (int sh = 1; sh <= 8; sh <<= 1) rs += __shfl_xor(rs, sh);
      lrow[v] = lrow[v] * corr + rs;
      mrow[v] = mnew;
      p0[v] = e0;
      p1[v] = e1;
#pragma unroll
      for (int t = 0; t < 4; ++t) o[t][v] *= corr;
    }

    // ---- transpose P (C layout) -> A layout via per-wave LDS tile ----
#pragma unroll
    for (int v = 0; v < 8; ++v) {
      int row = mo + v;
      pl[row * 32 + nl]      = f2bf(p0[v]);
      pl[row * 32 + 16 + nl] = f2bf(p1[v]);
    }
    asm volatile("s_wait_dscnt 0" ::: "memory");
    v16bf pf = load_frag(pl, 32);
    asm volatile("s_wait_dscnt 0" ::: "memory");

    // ---- O += P @ V : V^T is [D,L], column n of B is contiguous along L ----
    o[0] = WMMA(pf, vb0, o[0], false);
    o[1] = WMMA(pf, vb1, o[1], true);   // same A (P) as previous WMMA
    o[2] = WMMA(pf, vb2, o[2], true);
    o[3] = WMMA(pf, vb3, o[3], true);
  }

  // ---- normalize and store y as bf16 [B,L,C] for the projection GEMM ----
#pragma unroll
  for (int t = 0; t < 4; ++t) {
#pragma unroll
    for (int v = 0; v < 8; ++v) {
      int l = qbase + mo + v;
      float val = o[t][v] / lrow[v];
      ybp[(size_t)l * CCH + h * DD + t * 16 + nl] = f2bf(val);
    }
  }
}

// ---------------------------------------------------------------------------
// Output projection: out = yb[8192,1024] @ W_proj[1024,1024], fp32 store.
// Same double-buffered structure as the QKV GEMM.
// ---------------------------------------------------------------------------
__global__ __launch_bounds__(256) void k_proj_gemm(
    const bh16* __restrict__ yb, const bh16* __restrict__ wT,
    float* __restrict__ out) {
  const int wave = threadIdx.x >> 5;
  const int lane = threadIdx.x & 31;
  const int mrow0 = (blockIdx.y * 8 + wave) * 16;
  const int ntb = blockIdx.x * 64;

  const bh16* ap = yb + (size_t)mrow0 * CCH;
  const bh16* bp = wT + (size_t)ntb * CCH;

  v8f acc[4];
#pragma unroll
  for (int t = 0; t < 4; ++t) acc[t] = vzero8();

  v16bf a  = load_frag(ap, CCH);
  v16bf b0 = load_frag(bp + 0 * 16 * CCH, CCH);
  v16bf b1 = load_frag(bp + 1 * 16 * CCH, CCH);
  v16bf b2 = load_frag(bp + 2 * 16 * CCH, CCH);
  v16bf b3 = load_frag(bp + 3 * 16 * CCH, CCH);

  for (int kk = 32; kk < CCH; kk += 32) {
    v16bf an  = load_frag(ap + kk, CCH);
    v16bf bn0 = load_frag(bp + 0 * 16 * CCH + kk, CCH);
    v16bf bn1 = load_frag(bp + 1 * 16 * CCH + kk, CCH);
    v16bf bn2 = load_frag(bp + 2 * 16 * CCH + kk, CCH);
    v16bf bn3 = load_frag(bp + 3 * 16 * CCH + kk, CCH);

    acc[0] = WMMA(a, b0, acc[0], false);
    acc[1] = WMMA(a, b1, acc[1], true);
    acc[2] = WMMA(a, b2, acc[2], true);
    acc[3] = WMMA(a, b3, acc[3], true);

    a = an; b0 = bn0; b1 = bn1; b2 = bn2; b3 = bn3;
  }
  acc[0] = WMMA(a, b0, acc[0], false);
  acc[1] = WMMA(a, b1, acc[1], true);
  acc[2] = WMMA(a, b2, acc[2], true);
  acc[3] = WMMA(a, b3, acc[3], true);

  const int mo = (lane >> 4) * 8;
  const int nl = lane & 15;
#pragma unroll
  for (int t = 0; t < 4; ++t) {
#pragma unroll
    for (int v = 0; v < 8; ++v) {
      int row = mrow0 + mo + v;
      out[(size_t)row * CCH + ntb + t * 16 + nl] = acc[t][v];
    }
  }
}

// ---------------------------------------------------------------------------
// Host launcher
// ---------------------------------------------------------------------------
extern "C" void kernel_launch(void* const* d_in, const int* in_sizes, int n_in,
                              void* d_out, int out_size, void* d_ws, size_t ws_size,
                              hipStream_t stream) {
  const float* x      = (const float*)d_in[0];   // [4,2048,1024]
  const float* W_attn = (const float*)d_in[1];   // [1024,3072]
  const float* W_proj = (const float*)d_in[2];   // [1024,1024]
  float* out = (float*)d_out;                    // [4,2048,1024] fp32

  char* ws = (char*)d_ws;
  size_t off = 0;
  auto alloc = [&](size_t bytes) -> void* {
    void* p = ws + off;
    off = (off + bytes + 255) & ~(size_t)255;
    return p;
  };
  const size_t NTOK = (size_t)BB * LL;           // 8192
  bh16* xb  = (bh16*)alloc(NTOK * CCH * 2);            // 16 MB
  bh16* waT = (bh16*)alloc((size_t)3 * CCH * CCH * 2); // 6 MB (W_attn^T)
  bh16* wpT = (bh16*)alloc((size_t)CCH * CCH * 2);     // 2 MB (W_proj^T)
  bh16* Qb  = (bh16*)alloc(NTOK * CCH * 2);            // 16 MB [B,H,L,D]
  bh16* Kb  = (bh16*)alloc(NTOK * CCH * 2);            // 16 MB [B,H,L,D]
  bh16* Vt  = (bh16*)alloc(NTOK * CCH * 2);            // 16 MB [B,H,D,L]
  bh16* yb  = (bh16*)alloc(NTOK * CCH * 2);            // 16 MB [B,L,C]
  (void)in_sizes; (void)n_in; (void)out_size; (void)ws_size;

  k_cvt  <<<dim3(1024), dim3(256), 0, stream>>>(x, xb, (int)(NTOK * CCH));
  k_cvt_t<<<dim3(1024), dim3(256), 0, stream>>>(W_attn, waT, CCH, 3 * CCH);
  k_cvt_t<<<dim3(512),  dim3(256), 0, stream>>>(W_proj, wpT, CCH, CCH);

  // 8192x3072x1024 GEMM: grid = (3072/64 N-tiles, 8192/(16*8) M-tiles)
  k_qkv_gemm<<<dim3(48, 64), dim3(256), 0, stream>>>(xb, waT, Qb, Kb, Vt);

  // attention: grid = (L/64 q-tile groups, B*H heads), 4 waves/block
  k_attn<<<dim3(LL / 64, BB * HH), dim3(128), 0, stream>>>(Qb, Kb, Vt, yb);

  // 8192x1024x1024 GEMM
  k_proj_gemm<<<dim3(16, 64), dim3(256), 0, stream>>>(yb, wpT, out);
}
